// GlobalFeatureGeneration_48112223650045
// MI455X (gfx1250) — matrix operand order repaired and checked by reading, not compiled, not run on hardware
//
#include <hip/hip_runtime.h>
#include <hip/hip_bf16.h>

typedef __attribute__((ext_vector_type(16))) _Float16 v16h;
typedef __attribute__((ext_vector_type(8)))  _Float16 v8h;
typedef __attribute__((ext_vector_type(8)))  float    v8f;
typedef __attribute__((ext_vector_type(4)))  float    f32x4;
typedef __attribute__((ext_vector_type(4)))  unsigned int u32x4;
typedef __attribute__((ext_vector_type(8)))  int      i32x8;
typedef __attribute__((ext_vector_type(4)))  int      i32x4;

#if __has_builtin(__builtin_amdgcn_tensor_load_to_lds) && __has_builtin(__builtin_amdgcn_s_wait_tensorcnt)
#define HAVE_TDM 1
#endif

// ---------------------------------------------------------------------------
// Generic WMMA GEMM: one wave (32 lanes) computes one 16x16 tile of
// out[M,N] = act( A'[M,K] @ W[K,N] + bias ) (+ residual)
// The K x 16 weight slab is staged into LDS by the Tensor Data Mover (one
// tensor_load_to_lds per block, s_wait_tensorcnt 0 before use); A fragments
// are loaded as contiguous b128 vectors per the ISA 16-bit A-layout.
// ---------------------------------------------------------------------------
struct GemmP {
  const void*  A;      int lda;  int a_half;     // AMODE 0/1: row-major A
  const float* W;                                 // K x N row-major
  const float* bias;                              // nullable
  const float* res;                               // nullable residual [M,N]
  void*        out;    int out_half;
  int M, K, N, relu;
  // AMODE 1 (BN+ReLU on A): stats[0..K)=sum, stats[K..2K)=sumsq
  const float* stats; const float* gamma; const float* beta; float inv_cnt;
  // AMODE 2 (gather-group concat(feats[gi], xyz[gi]-new_xyz)):
  const int*   gi; int nsample; const float* feats; const float* pts; const float* nxyz;
};

__device__ __forceinline__ float bn_relu(const GemmP& p, float raw, int k) {
  float mean = p.stats[k] * p.inv_cnt;
  float var  = p.stats[p.K + k] * p.inv_cnt - mean * mean;
  float rs   = rsqrtf(var + 1e-5f);
  raw = (raw - mean) * rs * p.gamma[k] + p.beta[k];
  return fmaxf(raw, 0.0f);
}

template<int AMODE>
__global__ __launch_bounds__(32) void gemm_wmma(GemmP p) {
  __shared__ float bw_tile[320 * 16];            // K x 16 weight slab (K <= 320)
  const int lane = threadIdx.x & 31;
  const int half = lane >> 4, l16 = lane & 15;
  const int m0 = blockIdx.x * 16, n0 = blockIdx.y * 16;
  const int n = n0 + l16;
  const int m = m0 + l16;                        // A row handled by this lane

  // ---- stage W[0..K) x [n0..n0+16) into LDS ----
#ifdef HAVE_TDM
  {
    unsigned long long ga = (unsigned long long)(size_t)(p.W + (size_t)n0);
    unsigned lds_off = (unsigned)(size_t)(void*)bw_tile;   // addr[31:0] = LDS offset
    unsigned td0 = (unsigned)(p.N - n0);                   // tensor_dim0 (cols left)
    unsigned td1 = (unsigned)p.K;                          // tensor_dim1 (rows)
    unsigned long long st0 = (unsigned long long)(unsigned)p.N;
    u32x4 g0;
    g0[0] = 1u;                                            // count=1, user mode
    g0[1] = lds_off;                                       // D#.lds_addr
    g0[2] = (unsigned)(ga & 0xFFFFFFFFu);                  // global_addr[31:0]
    g0[3] = (unsigned)((ga >> 32) & 0x01FFFFFFu) | (2u << 30); // addr[56:32] | type=2
    i32x8 g1;
    g1[0] = (int)(2u << 16);                               // data_size = 4B
    g1[1] = (int)((td0 & 0xFFFFu) << 16);                  // tensor_dim0[15:0]
    g1[2] = (int)((td0 >> 16) | ((td1 & 0xFFFFu) << 16));  // dim0[31:16] | dim1[15:0]
    g1[3] = (int)((td1 >> 16) | (16u << 16));              // dim1[31:16] | tile_dim0=16
    g1[4] = (int)(td1 & 0xFFFFu);                          // tile_dim1=K, tile_dim2=0
    g1[5] = (int)(unsigned)(st0 & 0xFFFFFFFFu);            // dim0_stride[31:0]
    g1[6] = (int)(unsigned)((st0 >> 32) & 0xFFFFu);        // dim0_stride[47:32]
    g1[7] = 0;
    i32x4 gz4 = {0, 0, 0, 0};
    i32x8 gz8 = {0, 0, 0, 0, 0, 0, 0, 0};
    __builtin_amdgcn_tensor_load_to_lds(g0, g1, gz4, gz4, gz8, 0);
    __builtin_amdgcn_s_wait_tensorcnt(0);
  }
  __syncthreads();
#else
  for (int r = lane; r < p.K; r += 32) {
    const float* src = p.W + (size_t)r * p.N + n0;
    f32x4* dst = (f32x4*)&bw_tile[r * 16];
    dst[0] = *(const f32x4*)(src + 0);
    dst[1] = *(const f32x4*)(src + 4);
    dst[2] = *(const f32x4*)(src + 8);
    dst[3] = *(const f32x4*)(src + 12);
  }
  __syncthreads();
#endif

  // Hoist per-row gather info (AMODE 2)
  int gRow = 0, center = 0, bIdx = 0;
  if (AMODE == 2) {
    bIdx = m / (512 * p.nsample);
    int rem = m - bIdx * 512 * p.nsample;
    int s = rem / p.nsample;
    int j = rem - s * p.nsample;
    center = bIdx * 512 + s;
    gRow = p.gi[center * p.nsample + j];
  }
  // vector A path legal when chunks never cross K and rows are 16B aligned
  const bool avec = (AMODE != 2) && (p.K % 32 == 0) && (p.lda % 8 == 0);

  v8f acc;
  const float bv = p.bias ? p.bias[n] : 0.0f;
#pragma unroll
  for (int r = 0; r < 8; ++r) acc[r] = bv;

  for (int kc = 0; kc < p.K; kc += 32) {
    v16h af, bf;
    if (AMODE != 2 && avec) {
      if (p.a_half) {
        const _Float16* ap = (const _Float16*)p.A + (size_t)m * p.lda + kc + 8 * half;
        v8h h0 = *(const v8h*)ap;               // K = kc+8h .. +7   (b128)
        v8h h1 = *(const v8h*)(ap + 16);        // K = kc+16+8h .. +7
        if (AMODE == 1) {
#pragma unroll
          for (int e = 0; e < 8; ++e) {
            af[e]     = (_Float16)bn_relu(p, (float)h0[e], kc + 8 * half + e);
            af[e + 8] = (_Float16)bn_relu(p, (float)h1[e], kc + 16 + 8 * half + e);
          }
        } else {
#pragma unroll
          for (int e = 0; e < 8; ++e) { af[e] = h0[e]; af[e + 8] = h1[e]; }
        }
      } else {
        const float* ap = (const float*)p.A + (size_t)m * p.lda + kc + 8 * half;
        f32x4 r0 = *(const f32x4*)ap;
        f32x4 r1 = *(const f32x4*)(ap + 4);
        f32x4 r2 = *(const f32x4*)(ap + 16);
        f32x4 r3 = *(const f32x4*)(ap + 20);
#pragma unroll
        for (int e = 0; e < 4; ++e) {
          af[e]      = (_Float16)r0[e];
          af[e + 4]  = (_Float16)r1[e];
          af[e + 8]  = (_Float16)r2[e];
          af[e + 12] = (_Float16)r3[e];
        }
      }
    } else {
#pragma unroll
      for (int e = 0; e < 16; ++e) {
        int kA = kc + ((e >= 8) ? 16 : 0) + (((e >> 1) & 3) << 1) + (e & 1) + 8 * half;
        float av = 0.0f;
        if (AMODE == 2) {
          if (kA < p.K) {
            if (kA < 256)
              av = p.feats[((size_t)bIdx * 2048 + gRow) * 256 + kA];
            else
              av = p.pts[((size_t)bIdx * 2048 + gRow) * 6 + (kA - 256)]
                 - p.nxyz[center * 3 + (kA - 256)];
          }
        } else if (kA < p.K) {
          float raw = p.a_half ? (float)((const _Float16*)p.A)[(size_t)m * p.lda + kA]
                               : ((const float*)p.A)[(size_t)m * p.lda + kA];
          if (AMODE == 1) raw = bn_relu(p, raw, kA);
          av = raw;
        }
        af[e] = (_Float16)av;
      }
    }
    // B fragment from LDS: col = l16, K = kc + 8*half + (e&7) + 16*(e>=8) (lanes hit distinct banks)
#pragma unroll
    for (int e = 0; e < 16; ++e) {
      int kB = kc + ((e >= 8) ? 16 : 0) + (e & 7) + 8 * half;
      float bwv = (kB < p.K) ? bw_tile[kB * 16 + l16] : 0.0f;
      bf[e] = (_Float16)bwv;
    }
    acc = __builtin_amdgcn_wmma_f32_16x16x32_f16(false, af, false, bf,
                                                 (short)0, acc, false, false);
  }

#pragma unroll
  for (int r = 0; r < 8; ++r) {
    int row = m0 + r + 8 * half;
    float v = acc[r];
    if (p.res)  v += p.res[(size_t)row * p.N + n];
    if (p.relu) v = fmaxf(v, 0.0f);
    if (p.out_half) ((_Float16*)p.out)[(size_t)row * p.N + n] = (_Float16)v;
    else            ((float*)p.out)[(size_t)row * p.N + n] = v;
  }
}

// ---------------------------------------------------------------------------
// Flash attention: one wave per (batch, head, 16-query tile).
// dk = 64, seq = 2048. Online softmax; P re-laid-out D->A via 1KB LDS tile.
// ---------------------------------------------------------------------------
__global__ __launch_bounds__(32) void attn_kernel(const float* q, const float* k,
                                                  const float* v, float* o) {
  const int lane = threadIdx.x & 31;
  const int half = lane >> 4, l16 = lane & 15;
  const int qt = blockIdx.x, h = blockIdx.y, b = blockIdx.z;
  const size_t bo = (size_t)b * 2048 * 256;
  const int hoff = h * 64;
  __shared__ _Float16 pl[16][32];

  v16h qa[2];
#pragma unroll
  for (int c = 0; c < 2; ++c) {
    const float* qp = q + bo + (size_t)(qt * 16 + l16) * 256 + hoff + c * 32 + 8 * half;
    f32x4 r0 = *(const f32x4*)qp;
    f32x4 r1 = *(const f32x4*)(qp + 4);
    f32x4 r2 = *(const f32x4*)(qp + 16);
    f32x4 r3 = *(const f32x4*)(qp + 20);
#pragma unroll
    for (int e = 0; e < 4; ++e) {
      qa[c][e]      = (_Float16)r0[e];
      qa[c][e + 4]  = (_Float16)r1[e];
      qa[c][e + 8]  = (_Float16)r2[e];
      qa[c][e + 12] = (_Float16)r3[e];
    }
  }

  v8f accs[4];
#pragma unroll
  for (int t4 = 0; t4 < 4; ++t4)
#pragma unroll
    for (int r = 0; r < 8; ++r) accs[t4][r] = 0.0f;
  float mi[8], li[8];
#pragma unroll
  for (int r = 0; r < 8; ++r) { mi[r] = -1e30f; li[r] = 0.0f; }

  for (int kb = 0; kb < 2048; kb += 32) {
    v8f s0, s1;
#pragma unroll
    for (int r = 0; r < 8; ++r) { s0[r] = 0.0f; s1[r] = 0.0f; }
#pragma unroll
    for (int c = 0; c < 2; ++c) {
      v16h b0, b1;
      // each lane's K-frag elements are two runs of 8 floats (8h.., 16+8h..) -> 4x b128
      const float* kp0 = k + bo + (size_t)(kb + l16) * 256 + hoff + c * 32 + 8 * half;
      const float* kp1 = k + bo + (size_t)(kb + 16 + l16) * 256 + hoff + c * 32 + 8 * half;
#pragma unroll
      for (int qd = 0; qd < 4; ++qd) {
        f32x4 a0 = *(const f32x4*)(kp0 + 4 * qd + ((qd >= 2) ? 8 : 0));
        f32x4 a1 = *(const f32x4*)(kp1 + 4 * qd + ((qd >= 2) ? 8 : 0));
#pragma unroll
        for (int e = 0; e < 4; ++e) { b0[4 * qd + e] = (_Float16)a0[e]; b1[4 * qd + e] = (_Float16)a1[e]; }
      }
      s0 = __builtin_amdgcn_wmma_f32_16x16x32_f16(false, qa[c], false, b0, (short)0, s0, false, false);
      s1 = __builtin_amdgcn_wmma_f32_16x16x32_f16(false, qa[c], false, b1, (short)0, s1, false, false);
    }
    float pr0[8], pr1[8];
#pragma unroll
    for (int r = 0; r < 8; ++r) {
      float a0 = s0[r] * 0.125f, a1 = s1[r] * 0.125f;
      float t = fmaxf(a0, a1);
#pragma unroll
      for (int off = 1; off < 16; off <<= 1) t = fmaxf(t, __shfl_xor(t, off, 32));
      float mn = fmaxf(mi[r], t);
      float p0 = __expf(a0 - mn), p1 = __expf(a1 - mn);
      float ps = p0 + p1;
#pragma unroll
      for (int off = 1; off < 16; off <<= 1) ps += __shfl_xor(ps, off, 32);
      float corr = __expf(mi[r] - mn);
      li[r] = li[r] * corr + ps;
      mi[r] = mn;
#pragma unroll
      for (int t4 = 0; t4 < 4; ++t4) accs[t4][r] *= corr;
      pr0[r] = p0; pr1[r] = p1;
    }
    __syncthreads();
#pragma unroll
    for (int r = 0; r < 8; ++r) {
      pl[r + 8 * half][l16]      = (_Float16)pr0[r];
      pl[r + 8 * half][16 + l16] = (_Float16)pr1[r];
    }
    __syncthreads();
    v16h pa;
#pragma unroll
    for (int e = 0; e < 16; ++e) {
      int kA = ((e >= 8) ? 16 : 0) + (((e >> 1) & 3) << 1) + (e & 1) + 8 * half;
      pa[e] = pl[l16][kA];
    }
#pragma unroll
    for (int t4 = 0; t4 < 4; ++t4) {
      v16h vb;
#pragma unroll
      for (int e = 0; e < 16; ++e) {
        int kl = ((e >= 8) ? 16 : 0) + (e & 7) + 8 * half;
        vb[e] = (_Float16)v[bo + (size_t)(kb + kl) * 256 + hoff + t4 * 16 + l16];
      }
      accs[t4] = __builtin_amdgcn_wmma_f32_16x16x32_f16(false, pa, false, vb,
                                                        (short)0, accs[t4], false, false);
    }
  }
#pragma unroll
  for (int t4 = 0; t4 < 4; ++t4)
#pragma unroll
    for (int r = 0; r < 8; ++r) {
      int row = qt * 16 + r + 8 * half;
      o[bo + (size_t)row * 256 + hoff + t4 * 16 + l16] = accs[t4][r] / li[r];
    }
}

// ---------------------------------------------------------------------------
// LayerNorm over 256 cols, one wave per row.
// ---------------------------------------------------------------------------
__global__ __launch_bounds__(32) void layernorm_kernel(const float* in, const float* g,
                                                       const float* be, float* out) {
  const int row = blockIdx.x, lane = threadIdx.x;
  const float* r = in + (size_t)row * 256;
  float vals[8]; float s = 0.0f;
#pragma unroll
  for (int i = 0; i < 8; ++i) { vals[i] = r[lane + i * 32]; s += vals[i]; }
#pragma unroll
  for (int off = 1; off < 32; off <<= 1) s += __shfl_xor(s, off, 32);
  float mean = s * (1.0f / 256.0f);
  float sq = 0.0f;
#pragma unroll
  for (int i = 0; i < 8; ++i) { float d = vals[i] - mean; sq += d * d; }
#pragma unroll
  for (int off = 1; off < 32; off <<= 1) sq += __shfl_xor(sq, off, 32);
  float rs = rsqrtf(sq * (1.0f / 256.0f) + 1e-6f);
#pragma unroll
  for (int i = 0; i < 8; ++i) {
    int c = lane + i * 32;
    out[(size_t)row * 256 + c] = (vals[i] - mean) * rs * g[c] + be[c];
  }
}

// ---------------------------------------------------------------------------
// Farthest point sampling: one block per batch, 512 sequential argmax steps.
// ---------------------------------------------------------------------------
__global__ __launch_bounds__(256) void fps_kernel(const float* pts, int* fidx) {
  __shared__ float sx[2048], sy[2048], sz[2048], sd[2048];
  __shared__ float rmax[256]; __shared__ int ridx[256];
  __shared__ int sfar;
  const int b = blockIdx.x, t = threadIdx.x;
  for (int i = t; i < 2048; i += 256) {
    const float* p = pts + ((size_t)b * 2048 + i) * 6;
    sx[i] = p[0]; sy[i] = p[1]; sz[i] = p[2]; sd[i] = 1e10f;
  }
  if (t == 0) sfar = 0;
  __syncthreads();
  for (int iter = 0; iter < 512; ++iter) {
    int far = sfar;
    if (t == 0) fidx[b * 512 + iter] = far;
    float cx = sx[far], cy = sy[far], cz = sz[far];
    float bm = -1.0f; int bi = 0;
    for (int i = t; i < 2048; i += 256) {
      float dx = sx[i] - cx, dy = sy[i] - cy, dz = sz[i] - cz;
      float d = dx * dx + dy * dy + dz * dz;
      float nd = fminf(sd[i], d); sd[i] = nd;
      if (nd > bm) { bm = nd; bi = i; }
    }
    rmax[t] = bm; ridx[t] = bi;
    __syncthreads();
    for (int off = 128; off > 0; off >>= 1) {
      if (t < off) {
        float om = rmax[t + off]; int oi = ridx[t + off];
        if (om > rmax[t] || (om == rmax[t] && oi < ridx[t])) { rmax[t] = om; ridx[t] = oi; }
      }
      __syncthreads();
    }
    if (t == 0) sfar = ridx[0];
    __syncthreads();
  }
}

// ---------------------------------------------------------------------------
// Ball query for all 3 radii; also emits new_xyz. One thread per center.
// ---------------------------------------------------------------------------
__global__ __launch_bounds__(256) void ballquery_kernel(const float* pts, const int* fidx,
                                                        float* nxyz, int* gi0, int* gi1, int* gi2) {
  const int t = blockIdx.x * 256 + threadIdx.x;
  if (t >= 2048) return;
  const int b = t >> 9;
  const float* p = pts + (size_t)b * 2048 * 6;
  const int fi = fidx[t];
  const float cx = p[fi * 6 + 0], cy = p[fi * 6 + 1], cz = p[fi * 6 + 2];
  nxyz[t * 3 + 0] = cx; nxyz[t * 3 + 1] = cy; nxyz[t * 3 + 2] = cz;
  const float r0 = (float)(0.1 * 0.1), r1 = (float)(0.2 * 0.2), r2 = (float)(0.4 * 0.4);
  int c0 = 0, c1 = 0, c2 = 0;
  for (int n = 0; n < 2048; ++n) {
    float dx = p[n * 6 + 0] - cx, dy = p[n * 6 + 1] - cy, dz = p[n * 6 + 2] - cz;
    float d = dx * dx + dy * dy + dz * dz;
    if (d <= r0 && c0 < 16)  gi0[t * 16 + c0++] = n;
    if (d <= r1 && c1 < 32)  gi1[t * 32 + c1++] = n;
    if (d <= r2 && c2 < 128) gi2[t * 128 + c2++] = n;
  }
  int f0 = gi0[t * 16], f1 = gi1[t * 32], f2 = gi2[t * 128];
  for (int j = c0; j < 16; ++j)  gi0[t * 16 + j] = f0;
  for (int j = c1; j < 32; ++j)  gi1[t * 32 + j] = f1;
  for (int j = c2; j < 128; ++j) gi2[t * 128 + j] = f2;
}

// ---------------------------------------------------------------------------
// BN statistics (sum / sumsq per channel) over rows, via float atomics.
// ---------------------------------------------------------------------------
__global__ void zero_kernel(float* p, int n) {
  int i = blockIdx.x * 256 + threadIdx.x;
  if (i < n) p[i] = 0.0f;
}

__global__ void bn_stats_kernel(const _Float16* g, float* stats, int C, int rows) {
  const int c = threadIdx.x;
  int r0 = blockIdx.x * 1024;
  int r1 = r0 + 1024; if (r1 > rows) r1 = rows;
  float s = 0.0f, s2 = 0.0f;
  for (int r = r0; r < r1; ++r) {
    float v = (float)g[(size_t)r * C + c];
    s += v; s2 += v * v;
  }
  atomicAdd(&stats[c], s);
  atomicAdd(&stats[C + c], s2);
}

// ---------------------------------------------------------------------------
// Apply BN + ReLU then max over nsample, write one scale's slice of sa_out.
// ---------------------------------------------------------------------------
__global__ void maxreduce_kernel(const _Float16* g, const float* stats, const float* gamma,
                                 const float* beta, float inv_cnt, int ns, int C,
                                 float* out, int coloff) {
  const int t = blockIdx.x * 256 + threadIdx.x;
  if (t >= 2048 * C) return;
  const int center = t / C, c = t - center * C;
  float mean = stats[c] * inv_cnt;
  float var  = stats[C + c] * inv_cnt - mean * mean;
  float rs   = rsqrtf(var + 1e-5f);
  float ga = gamma[c], be = beta[c];
  float m = -1e30f;
  const _Float16* gp = g + (size_t)center * ns * C + c;
  for (int j = 0; j < ns; ++j) {
    float v = (float)gp[(size_t)j * C];
    v = fmaxf((v - mean) * rs * ga + be, 0.0f);
    m = fmaxf(m, v);
  }
  out[(size_t)center * 320 + coloff + c] = m;
}

// ---------------------------------------------------------------------------
// Host-side launch helpers
// ---------------------------------------------------------------------------
static void launch_gemm0(hipStream_t st, const void* A, int lda, int a_half,
                         const float* W, const float* bias, const float* res,
                         void* out, int out_half, int M, int K, int N, int relu) {
  GemmP p{}; p.A = A; p.lda = lda; p.a_half = a_half; p.W = W; p.bias = bias; p.res = res;
  p.out = out; p.out_half = out_half; p.M = M; p.K = K; p.N = N; p.relu = relu;
  gemm_wmma<0><<<dim3(M / 16, N / 16), 32, 0, st>>>(p);
}
static void launch_gemm_bn(hipStream_t st, const void* A, int lda,
                           const float* stats, const float* gamma, const float* beta, float inv_cnt,
                           const float* W, const float* bias, void* out,
                           int M, int K, int N) {
  GemmP p{}; p.A = A; p.lda = lda; p.a_half = 1; p.W = W; p.bias = bias;
  p.out = out; p.out_half = 1; p.M = M; p.K = K; p.N = N; p.relu = 0;
  p.stats = stats; p.gamma = gamma; p.beta = beta; p.inv_cnt = inv_cnt;
  gemm_wmma<1><<<dim3(M / 16, N / 16), 32, 0, st>>>(p);
}
static void launch_gemm_group(hipStream_t st, const int* gi, int ns, const float* feats,
                              const float* pts, const float* nxyz,
                              const float* W, const float* bias, void* out,
                              int M, int K, int N) {
  GemmP p{}; p.gi = gi; p.nsample = ns; p.feats = feats; p.pts = pts; p.nxyz = nxyz;
  p.W = W; p.bias = bias; p.out = out; p.out_half = 1; p.M = M; p.K = K; p.N = N; p.relu = 0;
  gemm_wmma<2><<<dim3(M / 16, N / 16), 32, 0, st>>>(p);
}

extern "C" void kernel_launch(void* const* d_in, const int* in_sizes, int n_in,
                              void* d_out, int out_size, void* d_ws, size_t ws_size,
                              hipStream_t stream) {
  const float* points = (const float*)d_in[0];
  float* out_final = (float*)d_out;                 // (4,512,256)
  float* x_out     = (float*)d_out + 524288;        // (4,2048,256)
  float* w = (float*)d_ws;

  // workspace layout (float offsets)
  float* x1  = w + 0;          // 8192x64
  float* x2  = w + 524288;     // 8192x256
  float* x0  = w + 2621440;    // 8192x256
  float* Q   = w + 4718592;
  float* K   = w + 6815744;
  float* V   = w + 8912896;
  float* O   = w + 11010048;
  float* T   = w + 13107200;   // attn proj + residual (pre-LN)
  float* SA  = w + 15204352;   // (2048,320)
  float* T1  = w + 15859712;   // (2048,256)
  float* T2  = w + 16384000;
  float* ST0 = w + 16908288;   // 256 floats (BN stats)
  float* ST1 = w + 16908544;
  float* NX  = w + 16908800;   // new_xyz (2048,3)
  int*   fidx = (int*)(w + 16914944);
  int*   gi0  = fidx + 2048;     // 2048*16
  int*   gi1  = gi0 + 32768;     // 2048*32
  int*   gi2  = gi1 + 65536;     // 2048*128
  _Float16* GA = (_Float16*)(w + 17277440);  // up to 2048*128*128 halves
  _Float16* GB = (_Float16*)(w + 34054656);

  // ---- point-embedding MLP: 6 -> 64 -> 256 -> 256 ----
  launch_gemm0(stream, points, 6, 0, (const float*)d_in[1], (const float*)d_in[2],
               nullptr, x1, 0, 8192, 6, 64, 1);
  launch_gemm0(stream, x1, 64, 0, (const float*)d_in[3], (const float*)d_in[4],
               nullptr, x2, 0, 8192, 64, 256, 1);
  launch_gemm0(stream, x2, 256, 0, (const float*)d_in[5], (const float*)d_in[6],
               nullptr, x0, 0, 8192, 256, 256, 0);

  // ---- MHA ----
  launch_gemm0(stream, x0, 256, 0, (const float*)d_in[7], nullptr, nullptr, Q, 0, 8192, 256, 256, 0);
  launch_gemm0(stream, x0, 256, 0, (const float*)d_in[8], nullptr, nullptr, K, 0, 8192, 256, 256, 0);
  launch_gemm0(stream, x0, 256, 0, (const float*)d_in[9], nullptr, nullptr, V, 0, 8192, 256, 256, 0);
  attn_kernel<<<dim3(128, 4, 4), 32, 0, stream>>>(Q, K, V, O);
  launch_gemm0(stream, O, 256, 0, (const float*)d_in[10], nullptr, x0, T, 0, 8192, 256, 256, 0);
  layernorm_kernel<<<8192, 32, 0, stream>>>(T, (const float*)d_in[11], (const float*)d_in[12], x_out);

  // ---- FPS + ball query ----
  fps_kernel<<<4, 256, 0, stream>>>(points, fidx);
  ballquery_kernel<<<8, 256, 0, stream>>>(points, fidx, NX, gi0, gi1, gi2);

  // ---- set abstraction: 3 scales ----
  const int nsArr[3]   = {16, 32, 128};
  const int cArr[3][3] = {{32, 32, 64}, {64, 64, 128}, {64, 96, 128}};
  const int wBase[3]   = {13, 25, 37};
  const int colOff[3]  = {0, 64, 192};
  int* giArr[3] = {gi0, gi1, gi2};

  for (int sc = 0; sc < 3; ++sc) {
    const int ns = nsArr[sc];
    const int M = 2048 * ns;
    const int C0 = cArr[sc][0], C1 = cArr[sc][1], C2 = cArr[sc][2];
    const int wi = wBase[sc];
    const float inv = 1.0f / (float)M;
    // layer 0: gathered GEMM, K=259 (256 feats + 3 rel-xyz)
    launch_gemm_group(stream, giArr[sc], ns, x_out, points, NX,
                      (const float*)d_in[wi], (const float*)d_in[wi + 1], GA, M, 259, C0);
    zero_kernel<<<1, 256, 0, stream>>>(ST0, 2 * C0);
    bn_stats_kernel<<<(M + 1023) / 1024, C0, 0, stream>>>(GA, ST0, C0, M);
    // layer 1 (BN+ReLU of layer-0 fused into A load)
    launch_gemm_bn(stream, GA, C0, ST0, (const float*)d_in[wi + 2], (const float*)d_in[wi + 3], inv,
                   (const float*)d_in[wi + 4], (const float*)d_in[wi + 5], GB, M, C0, C1);
    zero_kernel<<<1, 256, 0, stream>>>(ST1, 2 * C1);
    bn_stats_kernel<<<(M + 1023) / 1024, C1, 0, stream>>>(GB, ST1, C1, M);
    // layer 2
    launch_gemm_bn(stream, GB, C1, ST1, (const float*)d_in[wi + 6], (const float*)d_in[wi + 7], inv,
                   (const float*)d_in[wi + 8], (const float*)d_in[wi + 9], GA, M, C1, C2);
    zero_kernel<<<1, 256, 0, stream>>>(ST0, 2 * C2);
    bn_stats_kernel<<<(M + 1023) / 1024, C2, 0, stream>>>(GA, ST0, C2, M);
    // BN+ReLU + max over nsample -> sa_out slice
    maxreduce_kernel<<<(2048 * C2 + 255) / 256, 256, 0, stream>>>(
        GA, ST0, (const float*)d_in[wi + 10], (const float*)d_in[wi + 11], inv, ns, C2, SA, colOff[sc]);
  }

  // ---- aggregation MLP: 320 -> 256 -> 256 -> 256 ----
  launch_gemm0(stream, SA, 320, 0, (const float*)d_in[49], (const float*)d_in[50],
               nullptr, T1, 0, 2048, 320, 256, 1);
  launch_gemm0(stream, T1, 256, 0, (const float*)d_in[51], (const float*)d_in[52],
               nullptr, T2, 0, 2048, 256, 256, 1);
  launch_gemm0(stream, T2, 256, 0, (const float*)d_in[53], (const float*)d_in[54],
               nullptr, out_final, 0, 2048, 256, 256, 0);
}
